// SelfAttention_48515950576119
// MI455X (gfx1250) — matrix operand order repaired and checked by reading, not compiled, not run on hardware
//
#include <hip/hip_runtime.h>
#include <hip/hip_bf16.h>
#include <math.h>

// ---------------------------------------------------------------------------
// Problem constants (fixed by the reference)
// ---------------------------------------------------------------------------
#define BB   4096
#define SS   31
#define CC   512
#define HH   8
#define HD   64            // CC / HH
#define MM   (BB * SS)     // 126976 rows, divisible by 64
#define EPS  1e-5f

typedef __bf16 bf16x16 __attribute__((ext_vector_type(16)));
typedef __bf16 bf16x8  __attribute__((ext_vector_type(8)));
typedef float  f32x8   __attribute__((ext_vector_type(8)));
typedef int    v4i     __attribute__((vector_size(16)));   // matches builtin proto

#if __has_builtin(__builtin_amdgcn_global_load_async_to_lds_b128) && \
    __has_builtin(__builtin_amdgcn_s_wait_asynccnt)
#define HAVE_ASYNC_LDS 1
#else
#define HAVE_ASYNC_LDS 0
#endif

__device__ __forceinline__ f32x8 wmma_bf16(bf16x16 a, bf16x16 b, f32x8 c) {
    // D = A(16x32) * B(32x16) + C(16x16), bf16 in / f32 out
    return __builtin_amdgcn_wmma_f32_16x16x32_bf16(
        false, a, false, b, (short)0, c, false, false);
}

__device__ __forceinline__ bf16x16 join8(bf16x8 lo, bf16x8 hi) {
    return __builtin_shufflevector(lo, hi, 0, 1, 2, 3, 4, 5, 6, 7,
                                   8, 9, 10, 11, 12, 13, 14, 15);
}

// ---------------------------------------------------------------------------
// 1) BN over x: per-channel partial sums (deterministic two-pass)
// ---------------------------------------------------------------------------
__global__ __launch_bounds__(256) void bn1_partial_kernel(
    const float* __restrict__ x, float* __restrict__ p1) {
    const int chunk = blockIdx.x;              // 128 chunks of 992 rows
    const int r0 = chunk * (MM / 128);
    for (int c = threadIdx.x; c < CC; c += 256) {
        float s = 0.f, sq = 0.f;
        for (int i = 0; i < MM / 128; ++i) {
            float v = x[(size_t)(r0 + i) * CC + c];
            s += v; sq += v * v;
        }
        p1[chunk * CC + c] = s;
        p1[128 * CC + chunk * CC + c] = sq;
    }
}

__global__ __launch_bounds__(256) void bn1_final_kernel(
    const float* __restrict__ p1, const float* __restrict__ nw,
    const float* __restrict__ nb, float* __restrict__ sb1) {
    const int c = blockIdx.x * 256 + threadIdx.x;
    if (c >= CC) return;
    float s = 0.f, sq = 0.f;
    for (int i = 0; i < 128; ++i) {
        s  += p1[i * CC + c];
        sq += p1[128 * CC + i * CC + c];
    }
    const float invN = 1.0f / (float)MM;
    float mu  = s * invN;
    float var = sq * invN - mu * mu;
    float rstd = rsqrtf(var + EPS);
    float scale = nw[c] * rstd;
    sb1[c]      = scale;
    sb1[CC + c] = nb[c] - mu * scale;
}

// ---------------------------------------------------------------------------
// 2) Elementwise: xn = bf16(x * scale + bias) ; weight fp32 -> bf16
// ---------------------------------------------------------------------------
__global__ __launch_bounds__(256) void xnorm_kernel(
    const float* __restrict__ x, const float* __restrict__ sb1,
    __bf16* __restrict__ xn) {
    const size_t total = (size_t)MM * CC;
    for (size_t i = (size_t)blockIdx.x * 256 + threadIdx.x; i < total;
         i += (size_t)gridDim.x * 256) {
        int c = (int)(i & (CC - 1));
        xn[i] = (__bf16)(x[i] * sb1[c] + sb1[CC + c]);
    }
}

__global__ __launch_bounds__(256) void conv_bf16_kernel(
    const float* __restrict__ src, __bf16* __restrict__ dst, int n) {
    for (int i = blockIdx.x * 256 + threadIdx.x; i < n; i += gridDim.x * 256)
        dst[i] = (__bf16)src[i];
}

// ---------------------------------------------------------------------------
// 3) WMMA GEMM: Y[m,n] = sum_k A[m,k] * W[n,k]   (A: MMxCC bf16, W: CCxCC bf16)
//    Block tile 64x64, 8 waves; wave -> 1 M-subtile x 2 N-subtiles.
//    The block's full B tile (64 x 512 bf16 = 64 KB) is staged into LDS with
//    GLOBAL_LOAD_ASYNC_TO_LDS_B128 (ASYNCcnt path), removing the 4x redundant
//    B reads across waves; A streams from global (xn is L2-resident).
//    outb != null : store bf16. Else: d_out fp32 = acc + bias[n] + resid[m,n].
// ---------------------------------------------------------------------------
__global__ __launch_bounds__(256) void gemm_bf16_kernel(
    const __bf16* __restrict__ A, const __bf16* __restrict__ W,
    __bf16* __restrict__ outb, float* __restrict__ outf,
    const float* __restrict__ bias, const float* __restrict__ resid) {
    __shared__ __attribute__((aligned(128))) __bf16 sB[64 * CC];   // 64 KB

    // ---- cooperative stage of W rows [blockIdx.y*64, +64) x K=512 ----
    const __bf16* Wt = W + (size_t)(blockIdx.y * 64) * CC;
    const int t = threadIdx.x;
#if HAVE_ASYNC_LDS
#pragma unroll
    for (int i = 0; i < 16; ++i) {            // 4096 chunks of 16B, 256 thr
        int ofs = (i * 256 + t) * 8;          // element offset (8 bf16 = 16B)
        __builtin_amdgcn_global_load_async_to_lds_b128(
            (__attribute__((address_space(1))) v4i*)(Wt + ofs),
            (__attribute__((address_space(3))) v4i*)(sB + ofs), 0, 0);
    }
    __builtin_amdgcn_s_wait_asynccnt(0);
#else
#pragma unroll
    for (int i = 0; i < 16; ++i) {
        int ofs = (i * 256 + t) * 8;
        *(bf16x8*)(sB + ofs) = *(const bf16x8*)(Wt + ofs);
    }
#endif
    __syncthreads();

    const int lane = t & 31;
    const int wv   = t >> 5;
    const int l15  = lane & 15;
    const int mi   = wv & 3;              // M subtile 0..3
    const int ni0  = (wv >> 2) << 1;      // N subtiles {0,1} or {2,3}
    const int mrow = blockIdx.x * 64 + mi * 16 + l15;          // A row (M layout)
    const int kbA  = (lane < 16) ? 0 : 16;                     // A K split
    const int kbB  = (lane < 16) ? 0 : 16;                     // B K split
    const size_t arow = (size_t)mrow * CC;
    const __bf16* bp0 = sB + (size_t)(ni0 * 16 + l15) * CC + kbB;        // LDS
    const __bf16* bp1 = sB + (size_t)(ni0 * 16 + 16 + l15) * CC + kbB;   // LDS

    f32x8 acc0 = {}, acc1 = {};
#pragma unroll 4
    for (int k0 = 0; k0 < CC; k0 += 32) {
        bf16x8  alo = *(const bf16x8*)(A + arow + k0 + kbA);
        bf16x8  ahi = *(const bf16x8*)(A + arow + k0 + kbA + 8);
        bf16x16 a   = join8(alo, ahi);
        bf16x16 b0  = *(const bf16x16*)(bp0 + k0);
        bf16x16 b1  = *(const bf16x16*)(bp1 + k0);
        acc0 = wmma_bf16(a, b0, acc0);
        acc1 = wmma_bf16(a, b1, acc1);
    }
    // C/D layout: VGPR r -> row r (lanes 0-15) / r+8 (lanes 16-31), col = lane%16
    const int mbase = blockIdx.x * 64 + mi * 16 + ((lane < 16) ? 0 : 8);
    const int nc0   = blockIdx.y * 64 + ni0 * 16 + l15;
    if (outb) {
#pragma unroll
        for (int r = 0; r < 8; ++r) {
            size_t row = (size_t)(mbase + r) * CC;
            outb[row + nc0]      = (__bf16)acc0[r];
            outb[row + nc0 + 16] = (__bf16)acc1[r];
        }
    } else {
#pragma unroll
        for (int r = 0; r < 8; ++r) {
            size_t row = (size_t)(mbase + r) * CC;
            outf[row + nc0]      = acc0[r] + bias[nc0]      + resid[row + nc0];
            outf[row + nc0 + 16] = acc1[r] + bias[nc0 + 16] + resid[row + nc0 + 16];
        }
    }
}

// ---------------------------------------------------------------------------
// 4) energy[b,h,qs,ks] = sum_c q[b,qs,h,c] * k[b,ks,h,c]
//    One wave per (b,h): 32x32 padded output, K = HD = 64 (two WMMA steps).
// ---------------------------------------------------------------------------
__global__ __launch_bounds__(256) void energy_kernel(
    const __bf16* __restrict__ q, const __bf16* __restrict__ k,
    float* __restrict__ e) {
    const int lane = threadIdx.x & 31;
    const int wv   = threadIdx.x >> 5;
    const int idx  = blockIdx.x * 8 + wv;       // (b*8 + h), 0..32767
    const int b = idx >> 3, h = idx & 7;
    const int l15 = lane & 15;
    const int kbA = (lane < 16) ? 0 : 16;
    const int kbB = (lane < 16) ? 0 : 16;
    const __bf16* qb = q + (size_t)b * SS * CC + h * HD;
    const __bf16* kb = k + (size_t)b * SS * CC + h * HD;

    f32x8 acc[2][2] = {{{}, {}}, {{}, {}}};
#pragma unroll
    for (int k0 = 0; k0 < HD; k0 += 32) {
        bf16x16 a[2], bmat[2];
#pragma unroll
        for (int mi = 0; mi < 2; ++mi) {
            int qs = mi * 16 + l15;
            if (qs < SS) {
                const __bf16* rp = qb + (size_t)qs * CC + k0 + kbA;
                a[mi] = join8(*(const bf16x8*)rp, *(const bf16x8*)(rp + 8));
            } else {
                bf16x16 z = {}; a[mi] = z;
            }
        }
#pragma unroll
        for (int ni = 0; ni < 2; ++ni) {
            int ks = ni * 16 + l15;
            if (ks < SS) {
                bmat[ni] = *(const bf16x16*)(kb + (size_t)ks * CC + k0 + kbB);
            } else {
                bf16x16 z = {}; bmat[ni] = z;
            }
        }
#pragma unroll
        for (int mi = 0; mi < 2; ++mi)
#pragma unroll
            for (int ni = 0; ni < 2; ++ni)
                acc[mi][ni] = wmma_bf16(a[mi], bmat[ni], acc[mi][ni]);
    }
    float* eb = e + (size_t)idx * (SS * SS);
    const int rofs = (lane < 16) ? 0 : 8;
#pragma unroll
    for (int mi = 0; mi < 2; ++mi)
#pragma unroll
        for (int ni = 0; ni < 2; ++ni)
#pragma unroll
            for (int r = 0; r < 8; ++r) {
                int qs = mi * 16 + r + rofs;
                int ks = ni * 16 + l15;
                if (qs < SS && ks < SS) eb[qs * SS + ks] = acc[mi][ni][r];
            }
}

// ---------------------------------------------------------------------------
// 5) BN over energy: per-position stats over (B,H) = 32768 samples
// ---------------------------------------------------------------------------
__global__ __launch_bounds__(256) void bn2_partial_kernel(
    const float* __restrict__ e, float* __restrict__ p2) {
    const int chunk = blockIdx.x;               // 64 chunks of 512 samples
    const int s0 = chunk * 512;
    for (int p = threadIdx.x; p < SS * SS; p += 256) {
        float s = 0.f, sq = 0.f;
        for (int i = 0; i < 512; ++i) {
            float v = e[(size_t)(s0 + i) * (SS * SS) + p];
            s += v; sq += v * v;
        }
        p2[chunk * 1024 + p] = s;
        p2[64 * 1024 + chunk * 1024 + p] = sq;
    }
}

__global__ __launch_bounds__(256) void bn2_final_kernel(
    const float* __restrict__ p2, const float* __restrict__ pw,
    const float* __restrict__ pb, float* __restrict__ sb2) {
    const int p = blockIdx.x * 256 + threadIdx.x;
    if (p >= SS * SS) return;
    float s = 0.f, sq = 0.f;
    for (int i = 0; i < 64; ++i) {
        s  += p2[i * 1024 + p];
        sq += p2[64 * 1024 + i * 1024 + p];
    }
    const float invN = 1.0f / 32768.0f;
    float mu  = s * invN;
    float var = sq * invN - mu * mu;
    float rstd = rsqrtf(var + EPS);
    const float invSqrtC = 0.04419417382415922f;   // 1/sqrt(512)
    float a = pw[p] * rstd * invSqrtC;
    sb2[p]        = a;
    sb2[1024 + p] = (pb[p] - mu * pw[p] * rstd) * invSqrtC;
}

// ---------------------------------------------------------------------------
// 6) Fused energy-BN -> softmax -> attn @ v.  One wave per (b,h).
//    Lane-per-row softmax into LDS (32x32 zero-padded), then WMMA vs v.
// ---------------------------------------------------------------------------
__global__ __launch_bounds__(256) void attn_av_kernel(
    const float* __restrict__ e, const float* __restrict__ sb2,
    const __bf16* __restrict__ v, __bf16* __restrict__ ao) {
    __shared__ __attribute__((aligned(32))) __bf16 sA[8][32 * 32];
    const int lane = threadIdx.x & 31;
    const int wv   = threadIdx.x >> 5;
    const int idx  = blockIdx.x * 8 + wv;
    const int b = idx >> 3, h = idx & 7;
    const float* eb = e + (size_t)idx * (SS * SS);

    if (lane < SS) {
        float ev[SS];
        float mx = -3.4e38f;
        const int base = lane * SS;
        for (int j = 0; j < SS; ++j) {
            float t = eb[base + j] * sb2[base + j] + sb2[1024 + base + j];
            ev[j] = t;
            mx = fmaxf(mx, t);
        }
        float sum = 0.f;
        for (int j = 0; j < SS; ++j) { ev[j] = __expf(ev[j] - mx); sum += ev[j]; }
        float inv = 1.0f / sum;
        for (int j = 0; j < SS; ++j) sA[wv][lane * 32 + j] = (__bf16)(ev[j] * inv);
        sA[wv][lane * 32 + SS] = (__bf16)0.f;
    } else {   // lane 31: zero-fill padding row 31
        for (int j = 0; j < 32; ++j) sA[wv][31 * 32 + j] = (__bf16)0.f;
    }
    __syncthreads();

    const int l15 = lane & 15;
    const int kbA = (lane < 16) ? 0 : 16;
    const int kbB = (lane < 16) ? 0 : 16;
    bf16x16 a[2];
#pragma unroll
    for (int mi = 0; mi < 2; ++mi) {
        const __bf16* rp = &sA[wv][(mi * 16 + l15) * 32 + kbA];
        a[mi] = join8(*(const bf16x8*)rp, *(const bf16x8*)(rp + 8));
    }
    const __bf16* vb = v + (size_t)b * SS * CC + h * HD;
    f32x8 acc[2][4] = {{{}, {}, {}, {}}, {{}, {}, {}, {}}};
#pragma unroll
    for (int ni = 0; ni < 4; ++ni) {
        bf16x16 bmat;
#pragma unroll
        for (int i = 0; i < 16; ++i) {
            int kk = kbB + i;   // key index (K dim = S, one 32-wide step)
            bmat[i] = (kk < SS) ? vb[(size_t)kk * CC + ni * 16 + l15] : (__bf16)0.f;
        }
#pragma unroll
        for (int mi = 0; mi < 2; ++mi)
            acc[mi][ni] = wmma_bf16(a[mi], bmat, acc[mi][ni]);
    }
    __bf16* aob = ao + (size_t)b * SS * CC + h * HD;
    const int rofs = (lane < 16) ? 0 : 8;
#pragma unroll
    for (int mi = 0; mi < 2; ++mi)
#pragma unroll
        for (int ni = 0; ni < 4; ++ni)
#pragma unroll
            for (int r = 0; r < 8; ++r) {
                int qs = mi * 16 + r + rofs;
                if (qs < SS) aob[(size_t)qs * CC + ni * 16 + l15] = (__bf16)acc[mi][ni][r];
            }
}

// ---------------------------------------------------------------------------
// Launcher
// ---------------------------------------------------------------------------
extern "C" void kernel_launch(void* const* d_in, const int* in_sizes, int n_in,
                              void* d_out, int out_size, void* d_ws, size_t ws_size,
                              hipStream_t stream) {
    const float* x      = (const float*)d_in[0];
    const float* norm_w = (const float*)d_in[1];
    const float* norm_b = (const float*)d_in[2];
    const float* Wq     = (const float*)d_in[3];
    const float* Wk     = (const float*)d_in[4];
    const float* Wv     = (const float*)d_in[5];
    const float* Wout   = (const float*)d_in[6];
    const float* b_out  = (const float*)d_in[7];
    const float* pen_w  = (const float*)d_in[8];
    const float* pen_b  = (const float*)d_in[9];
    float* out = (float*)d_out;

    // ---- workspace layout (bytes, 256B-aligned regions) ----
    char* w = (char*)d_ws;
    size_t o = 0;
    float* p1  = (float*)(w + o); o += (size_t)128 * CC * 2 * 4;       // 512 KB
    float* sb1 = (float*)(w + o); o += (size_t)CC * 2 * 4;             // 4 KB
    float* p2  = (float*)(w + o); o += (size_t)64 * 1024 * 2 * 4;      // 512 KB
    float* sb2 = (float*)(w + o); o += (size_t)1024 * 2 * 4;           // 8 KB
    __bf16* wqb = (__bf16*)(w + o); o += (size_t)CC * CC * 2;
    __bf16* wkb = (__bf16*)(w + o); o += (size_t)CC * CC * 2;
    __bf16* wvb = (__bf16*)(w + o); o += (size_t)CC * CC * 2;
    __bf16* wob = (__bf16*)(w + o); o += (size_t)CC * CC * 2;
    __bf16* xn  = (__bf16*)(w + o);
    float*  energy = (float*)(w + o);             // overlays xn (xn dead after QKV)
    o += (size_t)MM * CC * 2;                     // 130 MB (>= energy's 126 MB)
    __bf16* qb = (__bf16*)(w + o);
    __bf16* attn_out = qb;                        // overlays q (q dead after energy)
    o += (size_t)MM * CC * 2;
    __bf16* kb = (__bf16*)(w + o); o += (size_t)MM * CC * 2;
    __bf16* vb = (__bf16*)(w + o); o += (size_t)MM * CC * 2;
    (void)ws_size; (void)in_sizes; (void)n_in; (void)out_size;

    // 1) BN(x) stats -> fused scale/bias
    bn1_partial_kernel<<<128, 256, 0, stream>>>(x, p1);
    bn1_final_kernel<<<2, 256, 0, stream>>>(p1, norm_w, norm_b, sb1);

    // 2) xn (bf16) and weights (bf16)
    xnorm_kernel<<<8192, 256, 0, stream>>>(x, sb1, xn);
    conv_bf16_kernel<<<256, 256, 0, stream>>>(Wq,   wqb, CC * CC);
    conv_bf16_kernel<<<256, 256, 0, stream>>>(Wk,   wkb, CC * CC);
    conv_bf16_kernel<<<256, 256, 0, stream>>>(Wv,   wvb, CC * CC);
    conv_bf16_kernel<<<256, 256, 0, stream>>>(Wout, wob, CC * CC);

    // 3) QKV projections (WMMA GEMMs, async-LDS staged B)
    dim3 ggrid(MM / 64, CC / 64);
    gemm_bf16_kernel<<<ggrid, 256, 0, stream>>>(xn, wqb, qb, nullptr, nullptr, nullptr);
    gemm_bf16_kernel<<<ggrid, 256, 0, stream>>>(xn, wkb, kb, nullptr, nullptr, nullptr);
    gemm_bf16_kernel<<<ggrid, 256, 0, stream>>>(xn, wvb, vb, nullptr, nullptr, nullptr);

    // 4) energy = q . k^T per (b,h)   (writes over xn region)
    energy_kernel<<<(BB * HH) / 8, 256, 0, stream>>>(qb, kb, energy);

    // 5) BN(energy) stats
    bn2_partial_kernel<<<64, 256, 0, stream>>>(energy, p2);
    bn2_final_kernel<<<4, 256, 0, stream>>>(p2, pen_w, pen_b, sb2);

    // 6) fused BN -> softmax -> attn@v   (writes over q region)
    attn_av_kernel<<<(BB * HH) / 8, 256, 0, stream>>>(energy, sb2, vb, attn_out);

    // 7) output projection + bias + residual -> fp32 d_out
    gemm_bf16_kernel<<<ggrid, 256, 0, stream>>>(attn_out, wob, nullptr, out, b_out, x);
}
